// Attention4D_77644418777285
// MI455X (gfx1250) — matrix operand order repaired and checked by reading, not compiled, not run on hardware
//
#include <hip/hip_runtime.h>

#define R7    7
#define NTOK  49
#define NH    8
#define KDIM  32
#define DHEAD 128
#define DHT   1024
#define DIMC  384

typedef __attribute__((ext_vector_type(16))) __bf16 v16bf;
typedef __attribute__((ext_vector_type(8)))  __bf16 bf16x8;
typedef __attribute__((ext_vector_type(8)))  float  v8f;
typedef __attribute__((ext_vector_type(4)))  float  f32x4;

#define Z8 ((v8f){0.f, 0.f, 0.f, 0.f, 0.f, 0.f, 0.f, 0.f})

static __device__ __forceinline__ v8f wmma_bf16(v16bf a, v16bf b, v8f c) {
  // D = A(16x32) * B(32x16) + C, f32 accumulate
  return __builtin_amdgcn_wmma_f32_16x16x32_bf16(false, a, false, b, (short)0, c,
                                                 false, false);
}

static __device__ __forceinline__ v16bf cat16(bf16x8 lo, bf16x8 hi) {
  return __builtin_shufflevector(lo, hi, 0, 1, 2, 3, 4, 5, 6, 7,
                                 8, 9, 10, 11, 12, 13, 14, 15);
}

// ---- A fragment (16x32 bf16) from row-major source (K contiguous), stride S ----
// lanes 0-15: row M=lane, K kb+0..7 (v0-3) & 16+kb..23+kb (v4-7); kb=8 for hi lanes.
static __device__ __forceinline__ v16bf ldfragA(const __bf16* A, int S, int r0, int c0) {
  const int lane = threadIdx.x & 31;
  const int r = r0 + (lane & 15);
  const int kb = (lane & 16) ? 8 : 0;
  const __bf16* p = A + r * S + c0 + kb;
  return cat16(*(const bf16x8*)p, *(const bf16x8*)(p + 16));
}

// ---- B fragment (32x16) from K-contiguous ("transposed") layout Bt[c*K + k] ----
// lanes 0-15: col N, K kb+0..7 & 16+kb..23+kb; kb=8 for hi lanes. Two 16B loads.
static __device__ __forceinline__ v16bf ldfragBt(const __bf16* Bt, int K, int k0, int c0) {
  const int lane = threadIdx.x & 31;
  const int c = c0 + (lane & 15);
  const int kb = (lane & 16) ? 8 : 0;
  const __bf16* p = Bt + c * K + k0 + kb;
  return cat16(*(const bf16x8*)p, *(const bf16x8*)(p + 16));
}

// A fragment from global f32 (x), row clamped, vectorized float4 reads + cvt
static __device__ __forceinline__ v16bf ldfragA_x(const float* A, int S, int r0, int c0,
                                                  int rmax) {
  const int lane = threadIdx.x & 31;
  int r = r0 + (lane & 15);
  if (r >= rmax) r = rmax - 1;   // duplicate last row; rows>=49 masked at store
  const int kb = (lane & 16) ? 8 : 0;
  const float* base = A + r * S + c0 + kb;
  const f32x4 a0 = *(const f32x4*)(base);
  const f32x4 a1 = *(const f32x4*)(base + 4);
  const f32x4 b0 = *(const f32x4*)(base + 16);
  const f32x4 b1 = *(const f32x4*)(base + 20);
  v16bf a;
#pragma unroll
  for (int j = 0; j < 4; ++j) {
    a[j]      = (__bf16)a0[j];
    a[4 + j]  = (__bf16)a1[j];
    a[8 + j]  = (__bf16)b0[j];
    a[12 + j] = (__bf16)b1[j];
  }
  return a;
}

// ---------------------------------------------------------------------------
// Prep kernel: transpose + convert weights to bf16 K-contiguous layouts in ws.
//   qkvT [1536][384] : rows 0-255 = q cols, 256-511 = k cols, 512-1535 = v cols
//   projT [384][1024]
// ---------------------------------------------------------------------------
#define QKVT_ELEMS (1536 * 384)
#define PROJT_ELEMS (384 * 1024)
extern "C" __global__ void __launch_bounds__(256)
prep_weights(const float* __restrict__ qw, const float* __restrict__ kw,
             const float* __restrict__ vw, const float* __restrict__ pw,
             __bf16* __restrict__ qkvT, __bf16* __restrict__ projT) {
  const int i = blockIdx.x * 256 + threadIdx.x;
  if (i < QKVT_ELEMS) {
    const int n = i / 384, k = i % 384;
    float v;
    if (n < 256)      v = qw[k * 256 + n];
    else if (n < 512) v = kw[k * 256 + (n - 256)];
    else              v = vw[k * 1024 + (n - 512)];
    qkvT[i] = (__bf16)v;
  } else if (i < QKVT_ELEMS + PROJT_ELEMS) {
    const int j = i - QKVT_ELEMS;
    const int n = j / 1024, k = j % 1024;
    projT[j] = (__bf16)pw[k * 384 + n];
  }
}

// LDS plan (bytes), total = 327680 == 320 KiB:
//  [0,      65536): q (64x256 bf16) | k (64x256 bf16)  --aliased by--> attn_bf16 [8][64][64]
//  [65536, 196608): vT (1024x64 bf16, channel-major)
//  [196608,327680): attn_f32 [8][49][49] (76832 B)     --aliased by--> out bf16 (64x1024)
extern "C" __global__ void __launch_bounds__(256, 1)
attn4d_fused(const float* __restrict__ x,
             const __bf16* __restrict__ qkvT, const __bf16* __restrict__ projT,
             const float* __restrict__ q_b, const float* __restrict__ q_g,
             const float* __restrict__ q_be,
             const float* __restrict__ k_b, const float* __restrict__ k_g,
             const float* __restrict__ k_be,
             const float* __restrict__ v_b, const float* __restrict__ v_g,
             const float* __restrict__ v_be,
             const float* __restrict__ vl_w, const float* __restrict__ vl_b,
             const float* __restrict__ vl_g, const float* __restrict__ vl_be,
             const float* __restrict__ th1_w, const float* __restrict__ th1_b,
             const float* __restrict__ th2_w, const float* __restrict__ th2_b,
             const float* __restrict__ p_b, const float* __restrict__ p_g,
             const float* __restrict__ p_be,
             const float* __restrict__ abias,
             float* __restrict__ out) {
  extern __shared__ char smem[];
  __bf16* qs    = (__bf16*)smem;                    // [64][256]
  __bf16* ksd   = qs + 64 * 256;                    // [64][256]
  __bf16* attnb = (__bf16*)smem;                    // [8][64][64] (aliases q/k)
  __bf16* vsT   = (__bf16*)(smem + 65536);          // [1024][64] channel-major
  float*  attnf = (float*)(smem + 196608);          // [8][49][49]
  __bf16* outs  = (__bf16*)(smem + 196608);         // [64][1024] (aliases attn_f32)

  const int tid  = threadIdx.x;
  const int wave = tid >> 5;
  const int lane = tid & 31;
  const float* xg = x + (size_t)blockIdx.x * NTOK * DIMC;
  float* og = out + (size_t)blockIdx.x * NTOK * DIMC;
  const float SCALE = 0.17677669529663687f;   // 32^-0.5, folded into q

  // Keep the bf16 weight images warm in L2 (global_prefetch_b8)
  __builtin_prefetch(qkvT + tid * 2048, 0, 1);
  __builtin_prefetch(projT + tid * 1024, 0, 1);

  // ---------------- Phase 1: QKV = x @ [q_w|k_w|v_w], BN folded ----------------
  {
    const int mT  = wave & 3;            // row tile 0..3 (rows 16*mT..)
    const int nt0 = (wave >> 2) * 48;    // 96 col tiles of 1536 split 2 ways
    v16bf xa[12];                        // 12 K-steps of x A-fragments, reused 48x
#pragma unroll
    for (int ks = 0; ks < 12; ++ks)
      xa[ks] = ldfragA_x(xg, DIMC, mT * 16, ks * 32, NTOK);
    const int rb = mT * 16 + ((lane & 16) ? 8 : 0);

    auto qkv_store = [&](int c0, const v8f& acc) {
      if (c0 < 256) {                    // -> q (row-major, A-matrix of logits)
        const int c = c0 + (lane & 15);
        const float bb = q_b[c], gg = q_g[c], be = q_be[c];
#pragma unroll
        for (int i = 0; i < 8; ++i) {
          const int r = rb + i;
          const float vv = ((acc[i] + bb) * gg + be) * SCALE;
          qs[r * 256 + c] = (__bf16)(r < NTOK ? vv : 0.0f);
        }
      } else if (c0 < 512) {             // -> k (row-major, read transposed later)
        const int c = (c0 - 256) + (lane & 15);
        const float bb = k_b[c], gg = k_g[c], be = k_be[c];
#pragma unroll
        for (int i = 0; i < 8; ++i) {
          const int r = rb + i;
          const float vv = (acc[i] + bb) * gg + be;
          ksd[r * 256 + c] = (__bf16)(r < NTOK ? vv : 0.0f);
        }
      } else {                           // -> vT (channel-major, one 16B store)
        const int c = (c0 - 512) + (lane & 15);
        const float bb = v_b[c], gg = v_g[c], be = v_be[c];
        bf16x8 pk;
#pragma unroll
        for (int i = 0; i < 8; ++i) {
          const int r = rb + i;
          const float vv = (acc[i] + bb) * gg + be;
          pk[i] = (__bf16)(r < NTOK ? vv : 0.0f);  // zero rows>=49: K-pad for attn@v
        }
        *(bf16x8*)(vsT + c * 64 + rb) = pk;
      }
    };

    // two column tiles per iteration + 1-deep software pipeline on B fragments:
    // next K-step's 4 b128 loads are issued before the current WMMAs consume b0/b1.
    for (int nt = 0; nt < 48; nt += 2) {
      const int c0 = (nt0 + nt) * 16;    // uniform per wave; pair never straddles q/k/v
      v8f acc0 = Z8, acc1 = Z8;
      v16bf b0 = ldfragBt(qkvT, 384, 0, c0);
      v16bf b1 = ldfragBt(qkvT, 384, 0, c0 + 16);
#pragma unroll
      for (int ks = 0; ks < 12; ++ks) {
        v16bf n0 = b0, n1 = b1;
        if (ks < 11) {
          n0 = ldfragBt(qkvT, 384, (ks + 1) * 32, c0);
          n1 = ldfragBt(qkvT, 384, (ks + 1) * 32, c0 + 16);
        }
        acc0 = wmma_bf16(xa[ks], b0, acc0);
        acc1 = wmma_bf16(xa[ks], b1, acc1);
        b0 = n0;
        b1 = n1;
      }
      qkv_store(c0, acc0);
      qkv_store(c0 + 16, acc1);
    }
  }
  __syncthreads();

  // ---------------- Phase 2: logits = q @ k^T (+rel-pos bias), one head per wave ------
  {
    const int h = wave;
    v16bf qa[4];
#pragma unroll
    for (int m = 0; m < 4; ++m) qa[m] = ldfragA(qs, 256, m * 16, h * KDIM);
#pragma unroll
    for (int n2 = 0; n2 < 4; ++n2) {
      // B(kk,n) = k[n][h*32+kk]: K-contiguous per lane in ksd rows
      const v16bf kb = ldfragBt(ksd + h * KDIM, 256, 0, n2 * 16);
      const int col = n2 * 16 + (lane & 15);
      const int cx = col % 7, cy = col / 7;
#pragma unroll
      for (int m2 = 0; m2 < 4; ++m2) {
        v8f acc = Z8;
        acc = wmma_bf16(qa[m2], kb, acc);
        if (col < NTOK) {
          const int rb = m2 * 16 + ((lane & 16) ? 8 : 0);
#pragma unroll
          for (int i = 0; i < 8; ++i) {
            const int r = rb + i;
            if (r < NTOK) {
              const int rx = r % 7, ry = r / 7;
              const int dx = rx > cx ? rx - cx : cx - rx;
              const int dy = ry > cy ? ry - cy : cy - ry;
              attnf[h * 2401 + r * 49 + col] = acc[i] + abias[h * 49 + dx * 7 + dy];
            }
          }
        }
      }
    }
  }
  __syncthreads();

  // ---------------- Phase 3: talking-heads-1, softmax(m), talking-heads-2 --------------
  {
    // zero attn_bf16 region (q/k now dead) -> K-dim zero padding for m in [49,64)
    uint32_t* z = (uint32_t*)attnb;
    for (int i = tid; i < 16384; i += 256) z[i] = 0u;

    float o1[10][8];
#pragma unroll
    for (int it = 0; it < 10; ++it) {
      const int p = tid + it * 256;          // p = n*49 + m
      if (p < 2401) {
        float a[8];
#pragma unroll
        for (int h = 0; h < 8; ++h) a[h] = attnf[h * 2401 + p];
#pragma unroll
        for (int g = 0; g < 8; ++g) {
          float s = th1_b[g];
#pragma unroll
          for (int h = 0; h < 8; ++h) s += a[h] * th1_w[h * 8 + g];
          o1[it][g] = s;
        }
      }
    }
    __syncthreads();
#pragma unroll
    for (int it = 0; it < 10; ++it) {
      const int p = tid + it * 256;
      if (p < 2401) {
#pragma unroll
        for (int g = 0; g < 8; ++g) attnf[g * 2401 + p] = o1[it][g];
      }
    }
    __syncthreads();
    // softmax over keys m for each (g, n): 392 rows of length 49
    for (int rr = tid; rr < 392; rr += 256) {
      float* row = attnf + (rr / 49) * 2401 + (rr % 49) * 49;
      float mx = row[0];
      for (int m = 1; m < 49; ++m) mx = fmaxf(mx, row[m]);
      float s = 0.f;
      for (int m = 0; m < 49; ++m) { const float e = __expf(row[m] - mx); row[m] = e; s += e; }
      const float inv = 1.0f / s;
      for (int m = 0; m < 49; ++m) row[m] *= inv;
    }
    __syncthreads();
    // talking-heads-2, write bf16 A-matrix [8][64][64] (zero-padded in m)
    for (int p = tid; p < 2401; p += 256) {
      const int n = p / 49, m = p % 49;
      float a[8];
#pragma unroll
      for (int h = 0; h < 8; ++h) a[h] = attnf[h * 2401 + p];
#pragma unroll
      for (int g = 0; g < 8; ++g) {
        float s = th2_b[g];
#pragma unroll
        for (int h = 0; h < 8; ++h) s += a[h] * th2_w[h * 8 + g];
        attnb[g * 4096 + n * 64 + m] = (__bf16)s;
      }
    }
  }
  __syncthreads();

  // ---------------- Phase 4: o = attn @ v, + depthwise3x3(v) BN, relu -> out bf16 ------
  {
    const int mT = wave & 3;
    const int hb = (wave >> 2) * 4;      // 4 heads per wave-group
    const int rb = mT * 16 + ((lane & 16) ? 8 : 0);

    auto epi = [&](int c0, const v8f& acc) {
      const int c = c0 + (lane & 15);
      const __bf16* vcol = vsT + c * 64;
      float wv[9];
#pragma unroll
      for (int t9 = 0; t9 < 9; ++t9) wv[t9] = vl_w[t9 * DHT + c];
      const float vb = vl_b[c], vg = vl_g[c], vbe = vl_be[c];
#pragma unroll
      for (int i = 0; i < 8; ++i) {
        const int p = rb + i;
        if (p < NTOK) {
          const int y = p / 7, x0 = p % 7;
          float s = 0.f;
#pragma unroll
          for (int dy = -1; dy <= 1; ++dy) {
#pragma unroll
            for (int dx = -1; dx <= 1; ++dx) {
              const int yy = y + dy, xx = x0 + dx;
              if (yy >= 0 && yy < 7 && xx >= 0 && xx < 7)
                s += (float)vcol[yy * 7 + xx] * wv[(dy + 1) * 3 + (dx + 1)];
            }
          }
          const float vloc = (s + vb) * vg + vbe;
          outs[p * DHT + c] = (__bf16)fmaxf(acc[i] + vloc, 0.0f);
        }
      }
    };

#pragma unroll
    for (int hh = 0; hh < 4; ++hh) {
      const __bf16* ah = attnb + (hb + hh) * 4096;
      const v16bf a0 = ldfragA(ah, 64, mT * 16, 0);
      const v16bf a1 = ldfragA(ah, 64, mT * 16, 32);
      for (int ct = 0; ct < 8; ct += 2) {   // two column tiles per iteration
        const int c0a = (hb + hh) * DHEAD + ct * 16;
        const int c0b = c0a + 16;
        const v16bf b0a = ldfragBt(vsT, 64, 0, c0a);
        const v16bf b0b = ldfragBt(vsT, 64, 0, c0b);
        const v16bf b1a = ldfragBt(vsT, 64, 32, c0a);
        const v16bf b1b = ldfragBt(vsT, 64, 32, c0b);
        v8f acc0 = Z8, acc1 = Z8;
        acc0 = wmma_bf16(a0, b0a, acc0);
        acc1 = wmma_bf16(a0, b0b, acc1);
        acc0 = wmma_bf16(a1, b1a, acc0);
        acc1 = wmma_bf16(a1, b1b, acc1);
        epi(c0a, acc0);
        epi(c0b, acc1);
      }
    }
  }
  __syncthreads();

  // ---------------- Phase 5: final projection relu(out) @ proj_w, BN folded ------------
  {
    const int mT  = wave & 3;
    const int nt0 = (wave >> 2) * 12;    // 24 col tiles of 384 split 2 ways
    v8f acc[12];
#pragma unroll
    for (int nt = 0; nt < 12; ++nt) acc[nt] = Z8;
    for (int ksg = 0; ksg < 4; ++ksg) {
      v16bf af[8];
#pragma unroll
      for (int j = 0; j < 8; ++j)
        af[j] = ldfragA(outs, DHT, mT * 16, (ksg * 8 + j) * 32);
#pragma unroll
      for (int nt = 0; nt < 12; ++nt) {
        const int c0 = (nt0 + nt) * 16;
#pragma unroll
        for (int j = 0; j < 8; ++j)
          acc[nt] = wmma_bf16(af[j], ldfragBt(projT, 1024, (ksg * 8 + j) * 32, c0), acc[nt]);
      }
    }
    const int rb = mT * 16 + ((lane & 16) ? 8 : 0);
#pragma unroll
    for (int nt = 0; nt < 12; ++nt) {
      const int c = (nt0 + nt) * 16 + (lane & 15);
      const float pb = p_b[c], pg = p_g[c], pbe = p_be[c];
#pragma unroll
      for (int i = 0; i < 8; ++i) {
        const int p = rb + i;
        if (p < NTOK) og[p * DIMC + c] = (acc[nt][i] + pb) * pg + pbe;
      }
    }
  }
}

extern "C" void kernel_launch(void* const* d_in, const int* in_sizes, int n_in,
                              void* d_out, int out_size, void* d_ws, size_t ws_size,
                              hipStream_t stream) {
  (void)n_in; (void)ws_size; (void)out_size;
  const float* x     = (const float*)d_in[0];
  const float* q_w   = (const float*)d_in[1];
  const float* q_b   = (const float*)d_in[2];
  const float* q_g   = (const float*)d_in[3];
  const float* q_be  = (const float*)d_in[4];
  const float* k_w   = (const float*)d_in[5];
  const float* k_b   = (const float*)d_in[6];
  const float* k_g   = (const float*)d_in[7];
  const float* k_be  = (const float*)d_in[8];
  const float* v_w   = (const float*)d_in[9];
  const float* v_b   = (const float*)d_in[10];
  const float* v_g   = (const float*)d_in[11];
  const float* v_be  = (const float*)d_in[12];
  const float* vl_w  = (const float*)d_in[13];
  const float* vl_b  = (const float*)d_in[14];
  const float* vl_g  = (const float*)d_in[15];
  const float* vl_be = (const float*)d_in[16];
  const float* th1_w = (const float*)d_in[17];
  const float* th1_b = (const float*)d_in[18];
  const float* th2_w = (const float*)d_in[19];
  const float* th2_b = (const float*)d_in[20];
  const float* p_w   = (const float*)d_in[21];
  const float* p_b   = (const float*)d_in[22];
  const float* p_g   = (const float*)d_in[23];
  const float* p_be  = (const float*)d_in[24];
  const float* abias = (const float*)d_in[25];

  // Workspace: bf16 transposed weights (1,966,080 bytes)
  __bf16* qkvT  = (__bf16*)d_ws;
  __bf16* projT = qkvT + QKVT_ELEMS;

  const int prep_total = QKVT_ELEMS + PROJT_ELEMS;
  prep_weights<<<(prep_total + 255) / 256, 256, 0, stream>>>(q_w, k_w, v_w, p_w, qkvT, projT);

  const int B = in_sizes[0] / (NTOK * DIMC);
  const size_t shmem = 65536 + 131072 + 131072;   // 327680 B = 320 KiB
  (void)hipFuncSetAttribute((const void*)attn4d_fused,
                            hipFuncAttributeMaxDynamicSharedMemorySize, (int)shmem);
  attn4d_fused<<<B, 256, shmem, stream>>>(
      x, qkvT, projT, q_b, q_g, q_be, k_b, k_g, k_be, v_b, v_g, v_be,
      vl_w, vl_b, vl_g, vl_be, th1_w, th1_b, th2_w, th2_b,
      p_b, p_g, p_be, abias, (float*)d_out);
}
